// TopKGatingNetwork_81647328297258
// MI455X (gfx1250) — compile-verified, matrix operand order checked by reading, not run on hardware
//
#include <hip/hip_runtime.h>
#include <hip/hip_bf16.h>

typedef __attribute__((ext_vector_type(16))) __bf16 v16bf;
typedef __attribute__((ext_vector_type(8)))  float  v8f;

#define N_ROWS 16384
#define D_IN   2048
#define D_Q    256
#define K_EXP  64

// RNE f32->bf16 pack (one-time prepack kernels only)
__device__ __forceinline__ unsigned int pack_bf16_rne(float a, float b) {
    union { float f; unsigned int u; } ua, ub;
    ua.f = a; ub.f = b;
    unsigned int ra = ua.u + 0x7FFFu + ((ua.u >> 16) & 1u);
    unsigned int rb = ub.u + 0x7FFFu + ((ub.u >> 16) & 1u);
    return (ra >> 16) | (rb & 0xFFFF0000u);
}

__device__ __forceinline__ unsigned short f32_to_bf16(float a) {
    union { float f; unsigned int u; } ua;
    ua.f = a;
    unsigned int r = ua.u + 0x7FFFu + ((ua.u >> 16) & 1u);
    return (unsigned short)(r >> 16);
}

// Single v_perm_b32 truncating pack: result = { hi16(fy), hi16(fx) }
__device__ __forceinline__ unsigned int pack_bf16_trunc(float fx, float fy) {
    union { float f; unsigned int u; } u0, u1;
    u0.f = fx; u1.f = fy;
    return __builtin_amdgcn_perm(u1.u, u0.u, 0x07060302u);
}

union Frag {
    unsigned int u[8];
    uint4        q[2];
    v16bf        v;
};

// K-offset inside a 16x32 bf16 fragment for vgpr v, lane-half h (ISA 7.12.2)
__device__ __forceinline__ int frag_kbase(int v, int half) {
    return ((v < 4) ? (2 * v) : (16 + 2 * (v - 4))) + half * 8;
}

// ---- Prepack w_q (fp32 [2048,256]) into B-fragment order: [tileN][kb][lane][v] ----
__global__ __launch_bounds__(256)
void prepack_wq(const float* __restrict__ w_q, unsigned int* __restrict__ wpk) {
    const int idx   = blockIdx.x * blockDim.x + threadIdx.x;  // 0..262143
    const int v     = idx & 7;
    const int lane  = (idx >> 3) & 31;
    const int kb    = (idx >> 8) & 63;
    const int tileN = idx >> 14;
    const int half  = lane >> 4, ln = lane & 15;
    const int col   = tileN * 16 + ln;
    const int k     = kb * 32 + frag_kbase(v, half);
    wpk[idx] = pack_bf16_rne(w_q[(size_t)k * D_Q + col],
                             w_q[(size_t)(k + 1) * D_Q + col]);
}

// ---- Prepack keys (fp32 [64,256]) into B-fragment order: [t][kb][lane][v] ----
__global__ __launch_bounds__(256)
void prepack_keys(const float* __restrict__ keys, unsigned int* __restrict__ kpk) {
    const int idx  = blockIdx.x * blockDim.x + threadIdx.x;   // 0..8191
    const int v    = idx & 7;
    const int lane = (idx >> 3) & 31;
    const int kb   = (idx >> 8) & 7;
    const int t    = idx >> 11;                               // 0..3
    const int half = lane >> 4, ln = lane & 15;
    const int n    = t * 16 + ln;
    const int k    = kb * 32 + frag_kbase(v, half);
    const float2 f = *(const float2*)(keys + (size_t)n * D_Q + k);
    kpk[idx] = pack_bf16_rne(f.x, f.y);
}

// Build a 16x32 bf16 A fragment from an fp32 row: 4 b128 loads + 8 v_perm
__device__ __forceinline__ Frag load_a_frag(const float* __restrict__ xk, int half) {
    const int ho = half * 8;
    const float4 f0 = *(const float4*)(xk + ho);        // k = ho..ho+3   -> v0,v1
    const float4 f1 = *(const float4*)(xk + ho + 4);    // k = ho+4..+7   -> v2,v3
    const float4 f2 = *(const float4*)(xk + ho + 16);   // k = 16+ho..+3  -> v4,v5
    const float4 f3 = *(const float4*)(xk + ho + 20);   // k = 20+ho..+3  -> v6,v7
    Frag a;
    a.u[0] = pack_bf16_trunc(f0.x, f0.y);
    a.u[1] = pack_bf16_trunc(f0.z, f0.w);
    a.u[2] = pack_bf16_trunc(f1.x, f1.y);
    a.u[3] = pack_bf16_trunc(f1.z, f1.w);
    a.u[4] = pack_bf16_trunc(f2.x, f2.y);
    a.u[5] = pack_bf16_trunc(f2.z, f2.w);
    a.u[6] = pack_bf16_trunc(f3.x, f3.y);
    a.u[7] = pack_bf16_trunc(f3.z, f3.w);
    return a;
}

// Load the 8 B fragments (two b128 each) for k-block kb.
__device__ __forceinline__ void load_b8(Frag* __restrict__ b,
                                        const unsigned int* __restrict__ wbase,
                                        int kb) {
#pragma unroll
    for (int t = 0; t < 8; ++t) {
        const uint4* bp = (const uint4*)(wbase + (size_t)t * 16384 + kb * 256);
        b[t].q[0] = bp[0];
        b[t].q[1] = bp[1];
    }
}

// ---------------- Kernel 1: query = x @ w_q + b_q  (store bf16) ---------------
// One wave per (16-row M-tile, 8-of-16 N-tiles): 2048 waves, 8 v8f accumulators.
// Explicit double-buffered B fragments so loads land directly in WMMA operands.
__global__ __launch_bounds__(256)
void gemm1_query(const float* __restrict__ x, const unsigned int* __restrict__ wpk,
                 const float* __restrict__ b_q, unsigned short* __restrict__ qbf) {
    const int wave  = (blockIdx.x * blockDim.x + threadIdx.x) >> 5; // 0..2047
    const int lane  = threadIdx.x & 31;
    const int half  = lane >> 4;
    const int ln    = lane & 15;
    const int tileM = wave >> 1;          // 0..1023
    const int nHalf = wave & 1;           // N tiles [nHalf*8, nHalf*8+8)
    const int row   = tileM * 16 + ln;

    const float* xrow = x + (size_t)row * D_IN;
    // B fragments for tile t, k-block kb at: wpk + t*16384 + kb*256 + lane*8
    const unsigned int* wbase = wpk + (size_t)nHalf * 8 * 16384 + (size_t)lane * 8;

    v8f acc[8] = {{}, {}, {}, {}, {}, {}, {}, {}};
    Frag b0[8], b1[8];
    load_b8(b0, wbase, 0);

    for (int kb = 0; kb < 64; kb += 2) {
        // Fill the other buffer while WMMAs consume this one.
        load_b8(b1, wbase, kb + 1);
        if (kb + 2 < 64)
            __builtin_prefetch(xrow + (kb + 2) * 32 + half * 8, 0, 1);
        Frag a = load_a_frag(xrow + kb * 32, half);
#pragma unroll
        for (int t = 0; t < 8; ++t)
            acc[t] = __builtin_amdgcn_wmma_f32_16x16x32_bf16(
                false, a.v, false, b0[t].v, (short)0, acc[t], false, false);

        if (kb + 2 < 64)
            load_b8(b0, wbase, kb + 2);
        a = load_a_frag(xrow + (kb + 1) * 32, half);
#pragma unroll
        for (int t = 0; t < 8; ++t)
            acc[t] = __builtin_amdgcn_wmma_f32_16x16x32_bf16(
                false, a.v, false, b1[t].v, (short)0, acc[t], false, false);
    }

#pragma unroll
    for (int t = 0; t < 8; ++t) {
        const int col  = (nHalf * 8 + t) * 16 + ln;
        const float bias = b_q[col];
#pragma unroll
        for (int r = 0; r < 8; ++r) {
            const int m = tileM * 16 + r + half * 8;     // C/D layout
            qbf[(size_t)m * D_Q + col] = f32_to_bf16(acc[t][r] + bias);
        }
    }
}

// ------- Kernel 2: logits = q @ keys^T /16 + 0.1*noise; top-2 softmax scatter -------
__global__ __launch_bounds__(256)
void gemm2_router(const unsigned short* __restrict__ qbf,
                  const unsigned int* __restrict__ kpk,
                  const float* __restrict__ noise,
                  float* __restrict__ out) {
    const int wave  = (blockIdx.x * blockDim.x + threadIdx.x) >> 5; // tileM 0..1023
    const int lane  = threadIdx.x & 31;
    const int half  = lane >> 4;
    const int ln    = lane & 15;
    const int tileM = wave;
    const int rowA  = tileM * 16 + ln;

    const unsigned short* qrow  = qbf + (size_t)rowA * D_Q;
    const unsigned int*   kbase = kpk + (size_t)lane * 8;

    v8f acc[4] = {{}, {}, {}, {}};
    for (int kb = 0; kb < 8; ++kb) {
        Frag barr[4];
#pragma unroll
        for (int t = 0; t < 4; ++t) {
            const uint4* bp = (const uint4*)(kbase + (size_t)(t * 8 + kb) * 256);
            barr[t].q[0] = bp[0];
            barr[t].q[1] = bp[1];
        }
        // A fragment from bf16 query: lane's dwords are two contiguous runs of
        // 8 bf16 (one b128 each) at k-offsets ho and 16+ho.
        Frag a;
        const int ho = half * 8;
        *(uint4*)&a.u[0] = *(const uint4*)(qrow + kb * 32 + ho);
        *(uint4*)&a.u[4] = *(const uint4*)(qrow + kb * 32 + ho + 16);
#pragma unroll
        for (int t = 0; t < 4; ++t) {
            acc[t] = __builtin_amdgcn_wmma_f32_16x16x32_bf16(
                false, a.v, false, barr[t].v, (short)0, acc[t], false, false);
        }
    }

    // Per-row top-2 over 64 experts, 2-way softmax, dense scatter.
#pragma unroll
    for (int r = 0; r < 8; ++r) {
        const int row = tileM * 16 + r + half * 8;
        float v1 = -1e30f, v2 = -1e30f;
        int   i1 = -1,     i2 = -1;
#pragma unroll
        for (int t = 0; t < 4; ++t) {
            const int c = t * 16 + ln;
            const float val = acc[t][r] * 0.0625f
                            + noise[(size_t)row * K_EXP + c] * 0.1f;
            if (val > v1)      { v2 = v1; i2 = i1; v1 = val; i1 = c; }
            else if (val > v2) { v2 = val; i2 = c; }
        }
        // XOR butterfly within each 16-lane half (masks < 16 keep halves separate)
#pragma unroll
        for (int mask = 1; mask <= 8; mask <<= 1) {
            const float ov1 = __shfl_xor(v1, mask, 32);
            const int   oi1 = __shfl_xor(i1, mask, 32);
            const float ov2 = __shfl_xor(v2, mask, 32);
            const int   oi2 = __shfl_xor(i2, mask, 32);
            if (ov1 > v1) {
                const float nv2 = (v1 > ov2) ? v1 : ov2;
                const int   ni2 = (v1 > ov2) ? i1 : oi2;
                v1 = ov1; i1 = oi1; v2 = nv2; i2 = ni2;
            } else if (ov1 > v2) {
                v2 = ov1; i2 = oi1;
            }
        }
        const float e  = __expf(v2 - v1);   // v1 >= v2
        const float p1 = 1.0f / (1.0f + e);
        const float p2 = 1.0f - p1;
#pragma unroll
        for (int t = 0; t < 4; ++t) {
            const int c = t * 16 + ln;
            float g = 0.0f;
            if (c == i1)      g = p1;
            else if (c == i2) g = p2;
            out[(size_t)row * K_EXP + c] = g;
        }
        if (ln == 0) {
            float* oi = out + (size_t)N_ROWS * K_EXP + (size_t)row * 2;
            oi[0] = (float)i1;
            oi[1] = (float)i2;
        }
    }
}

extern "C" void kernel_launch(void* const* d_in, const int* in_sizes, int n_in,
                              void* d_out, int out_size, void* d_ws, size_t ws_size,
                              hipStream_t stream) {
    (void)in_sizes; (void)n_in; (void)out_size; (void)ws_size;
    const float* x     = (const float*)d_in[0];
    const float* noise = (const float*)d_in[1];
    const float* w_q   = (const float*)d_in[2];
    const float* b_q   = (const float*)d_in[3];
    const float* keys  = (const float*)d_in[4];
    float* out = (float*)d_out;

    unsigned short* qbf = (unsigned short*)d_ws;                        // 8 MB
    unsigned int*   wpk = (unsigned int*)((char*)d_ws + (8u << 20));    // 1 MB
    unsigned int*   kpk = (unsigned int*)((char*)d_ws + (9u << 20));    // 32 KB

    prepack_wq  <<<1024, 256, 0, stream>>>(w_q, wpk);   // 262144 dwords
    prepack_keys<<<  32, 256, 0, stream>>>(keys, kpk);  //   8192 dwords
    // 2048 waves (1024 M-tiles x 2 N-halves), 8 waves/block
    gemm1_query <<< 256, 256, 0, stream>>>(x, wpk, b_q, qbf);
    // 1024 row-tile waves, 8 waves/block
    gemm2_router<<< 128, 256, 0, stream>>>(qbf, kpk, noise, out);
}
